// ScalarDotProductCriticNetworkDualAttention_78649441124612
// MI455X (gfx1250) — matrix operand order repaired and checked by reading, not compiled, not run on hardware
//
#include <hip/hip_runtime.h>
#include <hip/hip_bf16.h>
#include <math.h>

// ---------------------------------------------------------------------------
// CDNA5 / gfx1250 implementation.  wave32, WMMA f32_16x16x32_f16 +
// global_load_async_to_lds_b128 (ASYNCcnt) for weight staging.
//
// Problem sizes (fixed): B=128, N=32, M=64, OBS=128, ACT=16, D_QK=128,
// D_AV=64, F_IN=144 (padded to 160 for K-tiling by 32).
//
// d_ws layout (halfs): transposed f16 weights [out][in_pad]
//   WkT   [128][128] @ 0
//   WqT   [128][128] @ 16384
//   WkoT  [128][128] @ 32768
//   WqoT  [128][128] @ 49152
//   WavT  [ 64][160] @ 65536   (cols 144..159 zero)
//   WavoT [ 64][160] @ 75776
//   Wf1T  [ 64][128] @ 86016
//   total 94208 halfs = 188416 bytes
// ---------------------------------------------------------------------------

typedef __attribute__((ext_vector_type(16))) _Float16 v16h;
typedef __attribute__((ext_vector_type(8)))  float    v8f;

#define OFF_WK    0
#define OFF_WQ    16384
#define OFF_WKO   32768
#define OFF_WQO   49152
#define OFF_WAV   65536
#define OFF_WAVO  75776
#define OFF_WF1   86016
#define WS_HALFS  94208

union Frag32B {
  v16h v;
  uint4 q[2];
};

// A fragment (16x32 f16, M x K), source row-major [M][lda] halfs.
// Layout used here (same k order as B): lane L: row M=L&15, K = kt*32 + 16*(L>=16) + e  (e=0..15 contiguous)
__device__ __forceinline__ v16h load_frag_A(const _Float16* p, int lda, int m0,
                                            int kt, int lane) {
  int m  = m0 + (lane & 15);
  int hi = (lane >> 4) & 1;
  int k  = kt * 32 + hi * 16;
  Frag32B f;
  f.q[0] = *(const uint4*)(p + m * lda + k);
  f.q[1] = *(const uint4*)(p + m * lda + k + 8);
  return f.v;
}

// B fragment (32x16 f16, K x N), source stored TRANSPOSED row-major [N][ldb]
// (i.e. element (k,n) at p[n*ldb + k]).
// Layout used here (same k order as A): lane L: col N=L&15, K = kt*32 + 16*(L>=16) + e  (e=0..15 contiguous)
__device__ __forceinline__ v16h load_frag_Bt(const _Float16* p, int ldb, int n0,
                                             int kt, int lane) {
  int n = n0 + (lane & 15);
  int k = kt * 32 + ((lane >> 4) & 1) * 16;
  Frag32B f;
  f.q[0] = *(const uint4*)(p + n * ldb + k);
  f.q[1] = *(const uint4*)(p + n * ldb + k + 8);
  return f.v;
}

// D layout (16x16 f32): lane L: col N=L&15, VGPR v: row M = v + 8*(L>=16)
#define D_LOOP(body)                                \
  {                                                 \
    int n_ = n0 + (lane & 15);                      \
    int mb_ = m0 + ((lane >> 4) & 1) * 8;           \
    _Pragma("unroll") for (int v = 0; v < 8; ++v) { \
      int m_ = mb_ + v;                             \
      body;                                         \
    }                                               \
  }

// Generic tiled WMMA GEMM: D[MT*16 x NT*16] = A[.. x KT*32] * B.
// `wave` is an SGPR (readfirstlane'd) so the tile loop is wave-uniform:
// scalar branch, EXEC stays all-ones for WMMA.
template <typename Sink>
__device__ __forceinline__ void gemm(const _Float16* A, int lda,
                                     const _Float16* Bt, int ldb, int MT, int NT,
                                     int KT, int wave, int lane, Sink sink) {
  for (int t = wave; t < MT * NT; t += 8) {
    int mt = t / NT;
    int nt = t - mt * NT;
    v8f acc = {};
    for (int kt = 0; kt < KT; ++kt) {
      v16h a = load_frag_A(A, lda, mt * 16, kt, lane);
      v16h b = load_frag_Bt(Bt, ldb, nt * 16, kt, lane);
      acc = __builtin_amdgcn_wmma_f32_16x16x32_f16(
          false, a, false, b, (short)0, acc, false, false);
    }
    sink(mt * 16, nt * 16, acc);
  }
}

// ---------------------------------------------------------------------------
// Kernel 0: convert + transpose shared weights into f16 workspace.
// ---------------------------------------------------------------------------
__global__ __launch_bounds__(256) void prep_weights_kernel(
    const float* __restrict__ Wk, const float* __restrict__ Wq,
    const float* __restrict__ Wko, const float* __restrict__ Wqo,
    const float* __restrict__ Wav, const float* __restrict__ Wavo,
    const float* __restrict__ Wf1, _Float16* __restrict__ out) {
  int idx = blockIdx.x * 256 + threadIdx.x;
  if (idx >= WS_HALFS) return;
  float v;
  if (idx < OFF_WAV) {  // four 128x128 QK projection weights
    int which = idx >> 14;
    const float* src = (which == 0) ? Wk : (which == 1) ? Wq
                       : (which == 2) ? Wko : Wqo;
    int r = idx & 16383;
    int o = r >> 7, i = r & 127;
    v = src[i * 128 + o];
  } else if (idx < OFF_WF1) {  // two [144->160][64] AV weights, transposed
    int r = idx - OFF_WAV;
    const float* src = (r < 10240) ? Wav : Wavo;
    int r2 = (r < 10240) ? r : r - 10240;
    int o = r2 / 160, i = r2 - o * 160;
    v = (i < 144) ? src[i * 64 + o] : 0.0f;
  } else {  // W_f1 [128][64] -> [64][128]
    int r = idx - OFF_WF1;
    int o = r >> 7, i = r & 127;
    v = Wf1[i * 64 + o];
  }
  out[idx] = (_Float16)v;
}

// ---------------------------------------------------------------------------
// Kernel 1: one workgroup (256 threads = 8 wave32) per batch element.
// ---------------------------------------------------------------------------
__global__ __launch_bounds__(256) void critic_dual_attention_kernel(
    const float* __restrict__ states,        // [128][32][128]
    const float* __restrict__ policies,      // [128][32][16]
    const float* __restrict__ actions,       // [128][32][16]
    const float* __restrict__ states_other,  // [128][64][128]
    const float* __restrict__ actions_other, // [128][64][16]
    const _Float16* __restrict__ wsh,        // converted weights
    const float* __restrict__ Wf2,           // [64][1]
    float* __restrict__ out_value,           // [128][32][32]
    float* __restrict__ out_w,               // [128][32][32]
    float* __restrict__ out_wo) {            // [128][32][64]
  __shared__ __align__(16) unsigned char smem[81920];
  const int b    = blockIdx.x;
  const int tid  = threadIdx.x;
  const int lane = tid & 31;
  const int wave = __builtin_amdgcn_readfirstlane(tid >> 5);  // SGPR
  const float qk_scale = 0.08838834764831845f;  // 1/sqrt(128)

  // Persistent LDS regions
  float*     sW    = (float*)(smem + 0);          // [32][32] scores -> softmax
  float*     sWo   = (float*)(smem + 4096);       // [32][64]
  _Float16*  sDiff = (_Float16*)(smem + 12288);   // [32][64]  AVp-AVa (f16)
  _Float16*  sCF   = (_Float16*)(smem + 16384);   // [32][128] [base | wav_other]
  unsigned char* scr = smem + 24576;              // 40960-byte phased arena
  _Float16*  sWf1  = (_Float16*)(smem + 65536);   // [64][128] async-staged Wf1T

  // -------- Kick off async copy of Wf1T (16KB) global -> LDS (ASYNCcnt). ----
  // Consumed only in phase 6; the DMA overlaps phases 1-5.
  {
    unsigned lds_base = (unsigned)(unsigned long long)sWf1;  // low 32b = LDS off
    unsigned long long g_base = (unsigned long long)(wsh + OFF_WF1);
#pragma unroll
    for (int i = 0; i < 4; ++i) {
      unsigned byte_off = (unsigned)((i * 256 + tid) * 16);
      unsigned lds_addr = lds_base + byte_off;
      unsigned long long gaddr = g_base + byte_off;
      asm volatile("global_load_async_to_lds_b128 %0, %1, off"
                   :: "v"(lds_addr), "v"(gaddr)
                   : "memory");
    }
  }

  // ---------------- Phase 1: Q, K, Qo projections + self scores -------------
  {
    _Float16* sQo = (_Float16*)(scr + 0);      // [32][128]
    _Float16* sS  = (_Float16*)(scr + 8192);   // [32][128]
    _Float16* sQ  = (_Float16*)(scr + 16384);  // [32][128]
    _Float16* sK  = (_Float16*)(scr + 24576);  // [32][128]
    for (int i = tid; i < 4096; i += 256)
      sS[i] = (_Float16)states[b * 4096 + i];
    __syncthreads();
    gemm(sS, 128, wsh + OFF_WQ, 128, 2, 8, 4, wave, lane,
         [&](int m0, int n0, const v8f& d) {
           D_LOOP(sQ[m_ * 128 + n_] = (_Float16)d[v]);
         });
    gemm(sS, 128, wsh + OFF_WK, 128, 2, 8, 4, wave, lane,
         [&](int m0, int n0, const v8f& d) {
           D_LOOP(sK[m_ * 128 + n_] = (_Float16)d[v]);
         });
    gemm(sS, 128, wsh + OFF_WQO, 128, 2, 8, 4, wave, lane,
         [&](int m0, int n0, const v8f& d) {
           D_LOOP(sQo[m_ * 128 + n_] = (_Float16)d[v]);
         });
    __syncthreads();
    // scores = (Q @ K^T) * scale   (B operand = K stored [j][d] == transposed)
    gemm(sQ, 128, sK, 128, 2, 2, 4, wave, lane,
         [&](int m0, int n0, const v8f& d) {
           D_LOOP(sW[m_ * 32 + n_] = d[v] * qk_scale);
         });
    __syncthreads();
  }

  // ---------------- Phase 2: other-branch keys + cross scores ---------------
  {
    _Float16* sQo = (_Float16*)(scr + 0);      // persists from phase 1
    _Float16* sSo = (_Float16*)(scr + 8192);   // [64][128]
    _Float16* sKo = (_Float16*)(scr + 24576);  // [64][128]
    for (int i = tid; i < 8192; i += 256)
      sSo[i] = (_Float16)states_other[b * 8192 + i];
    __syncthreads();
    gemm(sSo, 128, wsh + OFF_WKO, 128, 4, 8, 4, wave, lane,
         [&](int m0, int n0, const v8f& d) {
           D_LOOP(sKo[m_ * 128 + n_] = (_Float16)d[v]);
         });
    __syncthreads();
    gemm(sQo, 128, sKo, 128, 2, 4, 4, wave, lane,
         [&](int m0, int n0, const v8f& d) {
           D_LOOP(sWo[m_ * 64 + n_] = d[v] * qk_scale);
         });
    __syncthreads();
  }

  // ---------------- Phase 3: row softmax (f32) + emit weight outputs --------
  if (tid < 32) {
    int i = tid;
    float mx = -1e30f;
    for (int j = 0; j < 32; ++j) mx = fmaxf(mx, sW[i * 32 + j]);
    float s = 0.f;
    for (int j = 0; j < 32; ++j) {
      float e = __expf(sW[i * 32 + j] - mx);
      sW[i * 32 + j] = e;
      s += e;
    }
    float inv = 1.0f / s;
    for (int j = 0; j < 32; ++j) {
      float w = sW[i * 32 + j] * inv;
      sW[i * 32 + j] = w;
      out_w[b * 1024 + i * 32 + j] = w;
    }
  } else if (tid < 64) {
    int i = tid - 32;
    float mx = -1e30f;
    for (int j = 0; j < 64; ++j) mx = fmaxf(mx, sWo[i * 64 + j]);
    float s = 0.f;
    for (int j = 0; j < 64; ++j) {
      float e = __expf(sWo[i * 64 + j] - mx);
      sWo[i * 64 + j] = e;
      s += e;
    }
    float inv = 1.0f / s;
    for (int j = 0; j < 64; ++j) {
      float w = sWo[i * 64 + j] * inv;
      sWo[i * 64 + j] = w;
      out_wo[b * 2048 + i * 64 + j] = w;
    }
  }
  __syncthreads();

  // ---------------- Phase 4: AV_act / AV_pol, diff, base --------------------
  {
    _Float16* sOA  = (_Float16*)(scr + 0);      // [32][160]
    _Float16* sOP  = (_Float16*)(scr + 10240);  // [32][160]
    float*    sAVa = (float*)(scr + 20480);     // [32][64]
    float*    sAVp = (float*)(scr + 28672);     // [32][64]
    for (int i = tid; i < 32 * 160; i += 256) {
      int r = i / 160, c = i - r * 160;
      float va, vp;
      if (c < 128) {
        va = vp = states[b * 4096 + r * 128 + c];
      } else if (c < 144) {
        va = actions[b * 512 + r * 16 + (c - 128)];
        vp = policies[b * 512 + r * 16 + (c - 128)];
      } else {
        va = vp = 0.0f;
      }
      sOA[i] = (_Float16)va;
      sOP[i] = (_Float16)vp;
    }
    __syncthreads();
    gemm(sOA, 160, wsh + OFF_WAV, 160, 2, 4, 5, wave, lane,
         [&](int m0, int n0, const v8f& d) {
           D_LOOP(sAVa[m_ * 64 + n_] = tanhf(d[v]));
         });
    gemm(sOP, 160, wsh + OFF_WAV, 160, 2, 4, 5, wave, lane,
         [&](int m0, int n0, const v8f& d) {
           D_LOOP(sAVp[m_ * 64 + n_] = tanhf(d[v]));
         });
    __syncthreads();
    // diff (f16) and base = weight @ AV_act -> CF[:, 0:64] (f16)
    for (int i = tid; i < 2048; i += 256)
      sDiff[i] = (_Float16)(sAVp[i] - sAVa[i]);
    for (int idx = tid; idx < 2048; idx += 256) {
      int i = idx >> 6, d = idx & 63;
      float acc = 0.f;
      for (int k = 0; k < 32; ++k) acc += sW[i * 32 + k] * sAVa[k * 64 + d];
      sCF[i * 128 + d] = (_Float16)acc;
    }
    __syncthreads();
  }

  // ---------------- Phase 5: AV_other, wav_other -> CF[:, 64:128] -----------
  {
    _Float16* sOAo = (_Float16*)(scr + 0);      // [64][160]
    float*    sAVo = (float*)(scr + 20480);     // [64][64]
    for (int i = tid; i < 64 * 160; i += 256) {
      int r = i / 160, c = i - r * 160;
      float v;
      if (c < 128)      v = states_other[b * 8192 + r * 128 + c];
      else if (c < 144) v = actions_other[b * 1024 + r * 16 + (c - 128)];
      else              v = 0.0f;
      sOAo[i] = (_Float16)v;
    }
    __syncthreads();
    gemm(sOAo, 160, wsh + OFF_WAVO, 160, 4, 4, 5, wave, lane,
         [&](int m0, int n0, const v8f& d) {
           D_LOOP(sAVo[m_ * 64 + n_] = tanhf(d[v]));
         });
    __syncthreads();
    for (int idx = tid; idx < 2048; idx += 256) {
      int i = idx >> 6, d = idx & 63;
      float acc = 0.f;
      for (int k = 0; k < 64; ++k) acc += sWo[i * 64 + k] * sAVo[k * 64 + d];
      sCF[i * 128 + 64 + d] = (_Float16)acc;
    }
    // Drain the async Wf1T staging before phase 6 reads it from LDS.
    asm volatile("s_wait_asynccnt 0" ::: "memory");
    __syncthreads();
  }

  // ---------------- Phase 6: C = CF @ Wf1, P = diff @ Wf1[0:64] -------------
  float* sC   = (float*)(scr + 0);      // [32][64]
  float* sP   = (float*)(scr + 8192);   // [32][64]
  float* sWf2 = (float*)(scr + 16384);  // [64]
  if (tid < 64) sWf2[tid] = Wf2[tid];
  gemm(sCF, 128, sWf1, 128, 2, 4, 4, wave, lane,
       [&](int m0, int n0, const v8f& d) {
         D_LOOP(sC[m_ * 64 + n_] = d[v]);
       });
  gemm(sDiff, 64, sWf1, 128, 2, 4, 2, wave, lane,
       [&](int m0, int n0, const v8f& d) {
         D_LOOP(sP[m_ * 64 + n_] = d[v]);
       });
  __syncthreads();

  // ---------------- Phase 7: value[i][j] = lrelu(C[i]+w[i,j]P[j]) . Wf2 -----
  for (int idx = tid; idx < 1024; idx += 256) {
    int i = idx >> 5, j = idx & 31;
    float wij = sW[i * 32 + j];
    const float* Ci = sC + i * 64;
    const float* Pj = sP + j * 64;
    float acc = 0.f;
#pragma unroll 8
    for (int d = 0; d < 64; ++d) {
      float z = Ci[d] + wij * Pj[d];
      z = (z > 0.f) ? z : 0.01f * z;
      acc += z * sWf2[d];
    }
    out_value[b * 1024 + idx] = acc;
  }
}

// ---------------------------------------------------------------------------
extern "C" void kernel_launch(void* const* d_in, const int* in_sizes, int n_in,
                              void* d_out, int out_size, void* d_ws,
                              size_t ws_size, hipStream_t stream) {
  const float* states        = (const float*)d_in[0];
  const float* policies      = (const float*)d_in[1];
  const float* actions       = (const float*)d_in[2];
  const float* states_other  = (const float*)d_in[3];
  const float* actions_other = (const float*)d_in[4];
  const float* W_key         = (const float*)d_in[5];
  const float* W_query       = (const float*)d_in[6];
  const float* W_av          = (const float*)d_in[7];
  const float* W_key_other   = (const float*)d_in[8];
  const float* W_query_other = (const float*)d_in[9];
  const float* W_av_other    = (const float*)d_in[10];
  const float* W_f1          = (const float*)d_in[11];
  const float* W_f2          = (const float*)d_in[12];

  _Float16* wsh = (_Float16*)d_ws;

  float* out_value = (float*)d_out;            // [128*32*32]
  float* out_w     = out_value + 131072;       // [128*32*32]
  float* out_wo    = out_value + 262144;       // [128*32*64]

  prep_weights_kernel<<<(WS_HALFS + 255) / 256, 256, 0, stream>>>(
      W_key, W_query, W_key_other, W_query_other, W_av, W_av_other, W_f1, wsh);

  critic_dual_attention_kernel<<<128, 256, 0, stream>>>(
      states, policies, actions, states_other, actions_other, wsh, W_f2,
      out_value, out_w, out_wo);

  (void)in_sizes; (void)n_in; (void)out_size; (void)ws_size;
}
